// SimpleGNN_16363825398402
// MI455X (gfx1250) — matrix operand order repaired and checked by reading, not compiled, not run on hardware
//
#include <hip/hip_runtime.h>

// ---------------------------------------------------------------------------
// NNConv-style GNN layer fused for MI455X (gfx1250, wave32, WMMA).
//
//   P) prep_weights : swizzle nn_w / root / lin_w / nn_b-matrix into per-lane
//                     WMMA B layout
//   Z) zero_agg     : clear the f32 aggregation buffer
//   E) edge_kernel  : per 16-edge group -> 64x v_wmma_f32_16x16x32_f16 with
//                     inline-0 C (edge MLP), relu via max(w,-b) (no C splat),
//                     bias term folded in with 2 WMMAs vs the fixed nn_b
//                     matrix, fused x[src] matvec. B tiles staged in LDS in
//                     two 32KB phases shared by the 8 waves of the block.
//                     Scatter via global_atomic_add_f32 into L2-resident agg.
//   N) node_kernel  : 2x WMMA (x@root + agg + bias), in-register LayerNorm via
//                     shfl_xor over 16-lane halves, relu, 1x WMMA (h@lin_w)
// ---------------------------------------------------------------------------

typedef __attribute__((ext_vector_type(16))) _Float16 v16h;
typedef __attribute__((ext_vector_type(8)))  float    v8f;

#define N_NODES  50000
#define N_EDGES  200000
#define IN_C     32
#define HID      32
#define OUT_C    16
#define EDGE_DIM 8
#define EG       (N_EDGES / 16)      // 12500 edge groups
#define NG       (N_NODES / 16)      // 3125 node tiles
#define NTILES   ((IN_C * HID) / 16) // 64 N-tiles of the 1024-wide edge MLP
#define XS_PITCH 20                  // feature-major x[src] pitch (pad 4: 16B-aligned rows)

// B-matrix (KxN, f16) per-lane layout assumption (mirrors sparse-doc 64x16
// pattern): lane l holds column N = l%16; half j (0..15) holds K = 8*(l/16)+j+(j&8) (same order as A below).
// A-matrix (MxK, f16) layout per ISA table: lane l holds row M = l%16;
//   lanes 0-15 : j<8 -> K=j,    j>=8 -> K=16+(j-8)
//   lanes 16-31: j<8 -> K=8+j,  j>=8 -> K=24+(j-8)
// C/D (MxN f32): VGPR r, lane l -> M = r + 8*(l/16), N = l%16.

__global__ void prep_weights(const float* __restrict__ nn_w,
                             const float* __restrict__ root,
                             const float* __restrict__ lin_w,
                             const float* __restrict__ nn_b,
                             _Float16* __restrict__ nnwB,
                             _Float16* __restrict__ rootB,
                             _Float16* __restrict__ linB,
                             _Float16* __restrict__ nnbB) {
    const int l = threadIdx.x;        // 32 lanes
    const int b = blockIdx.x;         // 69 tiles: 64 nn_w, 2 root, 1 lin_w, 2 nn_b
    const int n = l & 15;
    const int kbase = (l >> 4) * 8;
    if (b < 64) {
        #pragma unroll
        for (int j = 0; j < 16; ++j) {
            const int k = kbase + j + (j & 8);
            const float v = (k < EDGE_DIM) ? nn_w[k * (IN_C * HID) + b * 16 + n] : 0.f;
            nnwB[(b * 32 + l) * 16 + j] = (_Float16)v;
        }
    } else if (b < 66) {
        const int t = b - 64;
        #pragma unroll
        for (int j = 0; j < 16; ++j)
            rootB[(t * 32 + l) * 16 + j] = (_Float16)root[(kbase + j + (j & 8)) * HID + t * 16 + n];
    } else if (b == 66) {
        #pragma unroll
        for (int j = 0; j < 16; ++j)
            linB[l * 16 + j] = (_Float16)lin_w[(kbase + j + (j & 8)) * OUT_C + n];
    } else {
        // nn_b reshaped as [IN_C][HID] matrix: bias term of the fused matvec.
        const int t = b - 67;
        #pragma unroll
        for (int j = 0; j < 16; ++j)
            nnbB[(t * 32 + l) * 16 + j] = (_Float16)nn_b[(kbase + j + (j & 8)) * HID + t * 16 + n];
    }
}

__global__ void zero_agg(float* __restrict__ agg, int count) {
    const int i = blockIdx.x * blockDim.x + threadIdx.x;
    if (i < count) agg[i] = 0.f;
}

__global__ __launch_bounds__(256) void edge_kernel(
    const float* __restrict__ x, const int* __restrict__ src,
    const int* __restrict__ dst, const float* __restrict__ edge_attr,
    const float* __restrict__ nn_b, const _Float16* __restrict__ nnwB,
    const _Float16* __restrict__ nnbB, float* __restrict__ agg) {
    // 32KB B-tile stage + 4KB (neg) bias + 8x2.5KB feature-major x[src] + idx
    __shared__ __align__(32) _Float16 Bs[32 * 512];      // 32 tiles per phase
    __shared__ float nnegb_s[NTILES * 16];               // -nn_b, 1024 floats
    __shared__ float xs[8][IN_C * XS_PITCH];             // [feature][edge], per wave
    __shared__ int   sidx[8][16];
    __shared__ int   didx[8][16];

    const int tid  = threadIdx.x;
    const int lane = tid & 31;
    const int w    = tid >> 5;
    const int g    = blockIdx.x * 8 + w;   // wave-uniform
    const bool active = (g < EG);          // no early return: all waves hit barriers
    const int half8 = (lane >> 4) << 3;
    const int o     = lane & 15;

    // Stage NEGATED nn_b once per block (used as the relu knee: max(w, -b)).
    for (int i = tid; i < NTILES * 16; i += 256) nnegb_s[i] = -nn_b[i];

    v16h a;
    #pragma unroll
    for (int j = 0; j < 16; ++j) a[j] = (_Float16)0.f;

    v8f acc0, acc1;   // msg accumulators in D layout: cols o and o+16
    #pragma unroll
    for (int r = 0; r < 8; ++r) { acc0[r] = 0.f; acc1[r] = 0.f; }

    if (active) {
        const int e0 = g * 16;
        if (lane < 16) {
            sidx[w][lane] = src[e0 + lane];
            didx[w][lane] = dst[e0 + lane];
        }
        __builtin_amdgcn_wave_barrier();
        asm volatile("" ::: "memory");
        // Gather x[src] feature-major: xs[feature][edge]; lane = feature.
        #pragma unroll 4
        for (int k = 0; k < 16; ++k)
            xs[w][lane * XS_PITCH + k] = x[(size_t)sidx[w][k] * IN_C + lane];
        // A matrix: edge_attr block 16x8, zero-padded to K=32 in f16 A layout.
        if (lane < 16) {
            const float4* ea = (const float4*)(edge_attr + (size_t)(e0 + lane) * EDGE_DIM);
            const float4 p0 = ea[0], p1 = ea[1];
            a[0] = (_Float16)p0.x; a[1] = (_Float16)p0.y;
            a[2] = (_Float16)p0.z; a[3] = (_Float16)p0.w;
            a[4] = (_Float16)p1.x; a[5] = (_Float16)p1.y;
            a[6] = (_Float16)p1.z; a[7] = (_Float16)p1.w;
        }
        __builtin_amdgcn_wave_barrier();
        asm volatile("" ::: "memory");

        // Bias term Sum_i x[e,i]*nn_b[i,o] -> initialize accumulators with
        // 2 WMMAs against the fixed nn_b matrix. A = x[src] tile in A layout.
        {
            const int m = lane & 15;
            const int kb = (lane >> 4) * 8;
            v16h ax;
            #pragma unroll
            for (int j = 0; j < 8; ++j)
                ax[j] = (_Float16)xs[w][(kb + j) * XS_PITCH + m];
            #pragma unroll
            for (int j = 0; j < 8; ++j)
                ax[8 + j] = (_Float16)xs[w][(16 + kb + j) * XS_PITCH + m];
            const v16h nb0 = *(const v16h*)(nnbB + (size_t)(0 * 32 + lane) * 16);
            const v16h nb1 = *(const v16h*)(nnbB + (size_t)(1 * 32 + lane) * 16);
            acc0 = __builtin_amdgcn_wmma_f32_16x16x32_f16(
                false, ax, false, nb0, (short)0, acc0, false, false);
            acc1 = __builtin_amdgcn_wmma_f32_16x16x32_f16(
                false, ax, false, nb1, (short)0, acc1, false, false);
        }
    }

    for (int phase = 0; phase < 2; ++phase) {
        __syncthreads();
        // Cooperative copy of 32 B tiles (32KB) into LDS: 128B per thread.
        {
            const uint4* srcB = (const uint4*)(nnwB + (size_t)phase * 32 * 512);
            uint4* dstB = (uint4*)Bs;
            #pragma unroll
            for (int i = 0; i < 8; ++i)
                dstB[tid + i * 256] = srcB[tid + i * 256];
        }
        __syncthreads();
        if (active) {
            #pragma unroll 2
            for (int i = 0; i < 16; ++i) {        // 16 tile-pairs per phase
                const int t0 = phase * 32 + 2 * i;
                const v16h b0 = *(const v16h*)(Bs + (2 * i)     * 512 + lane * 16);
                const v16h b1 = *(const v16h*)(Bs + (2 * i + 1) * 512 + lane * 16);
                const float negb0 = nnegb_s[t0 * 16 + o];       // = -nn_b
                const float negb1 = nnegb_s[t0 * 16 + 16 + o];
                // Inline-0 C operand: no 8-register bias splat needed.
                v8f zc0, zc1;
                #pragma unroll
                for (int r = 0; r < 8; ++r) { zc0[r] = 0.f; zc1[r] = 0.f; }
                v8f d0 = __builtin_amdgcn_wmma_f32_16x16x32_f16(
                    false, a, false, b0, (short)0, zc0, false, false);
                v8f d1 = __builtin_amdgcn_wmma_f32_16x16x32_f16(
                    false, a, false, b1, (short)0, zc1, false, false);
                // x[src_e][ii] for e = half8..half8+7: two aligned b128 LDS loads.
                const int ii = phase * 16 + i;
                const float4 xlo = *(const float4*)&xs[w][ii * XS_PITCH + half8];
                const float4 xhi = *(const float4*)&xs[w][ii * XS_PITCH + half8 + 4];
                float xr[8] = {xlo.x, xlo.y, xlo.z, xlo.w, xhi.x, xhi.y, xhi.z, xhi.w};
                // relu(w+b) = max(w,-b) + b; the +b part is already in acc.
                #pragma unroll
                for (int r = 0; r < 8; ++r) {
                    acc0[r] += xr[r] * fmaxf(d0[r], negb0);
                    acc1[r] += xr[r] * fmaxf(d1[r], negb1);
                }
            }
        }
    }

    if (active) {
        // Scatter-add msg into agg[dst]: stays in L2 (6.4 MB buffer).
        #pragma unroll
        for (int r = 0; r < 8; ++r) {
            const int node = didx[w][r + half8];
            atomicAdd(&agg[(size_t)node * HID + o],      acc0[r]);
            atomicAdd(&agg[(size_t)node * HID + 16 + o], acc1[r]);
        }
    }
}

__global__ __launch_bounds__(256) void node_kernel(
    const float* __restrict__ x, const float* __restrict__ agg,
    const float* __restrict__ bias, const float* __restrict__ norm_w,
    const float* __restrict__ norm_b, const float* __restrict__ lin_b,
    const _Float16* __restrict__ rootB, const _Float16* __restrict__ linB,
    float* __restrict__ out) {
    __shared__ float hbuf[8][16 * HID];   // per-wave h tile for layout re-entry

    const int lane = threadIdx.x & 31;
    const int w    = threadIdx.x >> 5;
    const int tile = blockIdx.x * 8 + w;  // wave-uniform
    if (tile >= NG) return;
    const int n0    = tile * 16;
    const int m     = lane & 15;
    const int hi    = lane >> 4;
    const int half8 = hi << 3;

    // A = x tile [16 x 32] in f16 A layout (K=32 exactly).
    const float* xrow = x + (size_t)(n0 + m) * IN_C + (hi ? 8 : 0);
    v16h a;
    #pragma unroll
    for (int j = 0; j < 8; ++j) a[j] = (_Float16)xrow[j];
    #pragma unroll
    for (int j = 0; j < 8; ++j) a[8 + j] = (_Float16)xrow[16 + j];

    // C = agg + bias (D layout).
    const float b0 = bias[m], b1 = bias[16 + m];
    v8f c0, c1;
    #pragma unroll
    for (int r = 0; r < 8; ++r) {
        const size_t node = (size_t)(n0 + r + half8);
        c0[r] = agg[node * HID + m]      + b0;
        c1[r] = agg[node * HID + 16 + m] + b1;
    }
    const v16h rb0 = *(const v16h*)(rootB + (size_t)(0 * 32 + lane) * 16);
    const v16h rb1 = *(const v16h*)(rootB + (size_t)(1 * 32 + lane) * 16);
    v8f d0 = __builtin_amdgcn_wmma_f32_16x16x32_f16(
        false, a, false, rb0, (short)0, c0, false, false);
    v8f d1 = __builtin_amdgcn_wmma_f32_16x16x32_f16(
        false, a, false, rb1, (short)0, c1, false, false);

    // LayerNorm over 32 features: shfl_xor masks 1,2,4,8 stay inside each
    // 16-lane half on wave32; combine the two N-tiles per lane first.
    const float nw0 = norm_w[m], nw1 = norm_w[16 + m];
    const float nb0 = norm_b[m], nb1 = norm_b[16 + m];
    #pragma unroll
    for (int r = 0; r < 8; ++r) {
        float s = d0[r] + d1[r];
        float q = d0[r] * d0[r] + d1[r] * d1[r];
        #pragma unroll
        for (int msk = 1; msk <= 8; msk <<= 1) {
            s += __shfl_xor(s, msk, 32);
            q += __shfl_xor(q, msk, 32);
        }
        const float mu  = s * (1.f / HID);
        const float var = q * (1.f / HID) - mu * mu;
        const float inv = rsqrtf(var + 1e-5f);
        const float h0  = fmaxf((d0[r] - mu) * inv * nw0 + nb0, 0.f);
        const float h1  = fmaxf((d1[r] - mu) * inv * nw1 + nb1, 0.f);
        hbuf[w][(r + half8) * HID + m]      = h0;
        hbuf[w][(r + half8) * HID + 16 + m] = h1;
    }
    __builtin_amdgcn_wave_barrier();
    asm volatile("" ::: "memory");

    // Re-enter A layout from LDS for the final Linear(32 -> 16).
    const float* hrow = &hbuf[w][m * HID + (hi ? 8 : 0)];
    v16h a2;
    #pragma unroll
    for (int j = 0; j < 8; ++j) a2[j] = (_Float16)hrow[j];
    #pragma unroll
    for (int j = 0; j < 8; ++j) a2[8 + j] = (_Float16)hrow[16 + j];
    const v16h lb = *(const v16h*)(linB + (size_t)lane * 16);
    const float lbv = lin_b[m];
    v8f c2;
    #pragma unroll
    for (int r = 0; r < 8; ++r) c2[r] = lbv;
    v8f d2 = __builtin_amdgcn_wmma_f32_16x16x32_f16(
        false, a2, false, lb, (short)0, c2, false, false);
    #pragma unroll
    for (int r = 0; r < 8; ++r)
        out[(size_t)(n0 + r + half8) * OUT_C + m] = d2[r];
}

extern "C" void kernel_launch(void* const* d_in, const int* in_sizes, int n_in,
                              void* d_out, int out_size, void* d_ws, size_t ws_size,
                              hipStream_t stream) {
    (void)in_sizes; (void)n_in; (void)out_size; (void)ws_size;
    const float* x         = (const float*)d_in[0];
    const int*   eidx      = (const int*)  d_in[1];   // [2, E]: row0=src, row1=dst
    const float* edge_attr = (const float*)d_in[2];
    const float* nn_w      = (const float*)d_in[3];
    const float* nn_b      = (const float*)d_in[4];
    const float* root      = (const float*)d_in[5];
    const float* bias      = (const float*)d_in[6];
    const float* norm_w    = (const float*)d_in[7];
    const float* norm_b    = (const float*)d_in[8];
    const float* lin_w     = (const float*)d_in[9];
    const float* lin_b     = (const float*)d_in[10];
    float* out = (float*)d_out;
    const int* src = eidx;
    const int* dst = eidx + N_EDGES;

    // Workspace: agg (6.4 MB f32) + prepped f16 weight tiles (~70 KB).
    char* ws = (char*)d_ws;
    float* agg = (float*)ws;
    const size_t aggBytes = (size_t)N_NODES * HID * sizeof(float); // 32B-aligned
    _Float16* nnwB  = (_Float16*)(ws + aggBytes);
    _Float16* rootB = nnwB + 64 * 512;
    _Float16* linB  = rootB + 2 * 512;
    _Float16* nnbB  = linB + 512;

    prep_weights<<<69, 32, 0, stream>>>(nn_w, root, lin_w, nn_b,
                                        nnwB, rootB, linB, nnbB);
    zero_agg<<<(N_NODES * HID + 255) / 256, 256, 0, stream>>>(agg, N_NODES * HID);
    edge_kernel<<<(EG + 7) / 8, 256, 0, stream>>>(x, src, dst, edge_attr, nn_b,
                                                  nnwB, nnbB, agg);
    node_kernel<<<(NG + 7) / 8, 256, 0, stream>>>(x, agg, bias, norm_w, norm_b,
                                                  lin_b, rootB, linB, out);
}
